// GATClassifier_75548474736836
// MI455X (gfx1250) — matrix operand order repaired and checked by reading, not compiled, not run on hardware
//
#include <hip/hip_runtime.h>
#include <hip/hip_bf16.h>
#include <math.h>

// ---- CDNA5 WMMA vector types (gfx1250, wave32) ----
typedef __attribute__((ext_vector_type(16))) __bf16 v16bf;
typedef __attribute__((ext_vector_type(8)))  float  v8f;

#define NNODES 50000
#define NEDGES 800000
#define NGRAPH 64
#define E2TOT  (NEDGES + NNODES)   // edges + self loops
#define KDIM   256                 // all GEMMs have K=256

// ---------------- device helpers ----------------
__device__ __forceinline__ unsigned short f32_to_bf16_rne(float f) {
    unsigned int u = __float_as_uint(f);
    unsigned int r = u + 0x7FFFu + ((u >> 16) & 1u);
    return (unsigned short)(r >> 16);
}
// order-preserving float -> uint key (for atomicMax over signed floats)
__device__ __forceinline__ unsigned int fkey(float f) {
    unsigned int u = __float_as_uint(f);
    return (u & 0x80000000u) ? ~u : (u | 0x80000000u);
}
__device__ __forceinline__ float fkey_dec(unsigned int k) {
    unsigned int u = (k & 0x80000000u) ? (k & 0x7FFFFFFFu) : ~k;
    return __uint_as_float(u);
}
// A-fragment swizzle inside a 32-wide K chunk: swap sub-blocks [8..15]<->[16..23]
// (ISA 7.12.2: lanes 0-15 hold K {0-7,16-23}; lanes 16-31 hold K {8-15,24-31})
__device__ __forceinline__ int apos(int r) {
    return (r & 7) | (((r >> 3) & 1) << 4) | (((r >> 4) & 1) << 3);
}

// ---------------- kernels ----------------
__global__ void k_zero(unsigned int* p, long long n) {
    long long i = blockIdx.x * (long long)blockDim.x + threadIdx.x;
    if (i < n) p[i] = 0u;
}

// f32 [rows,256] -> bf16 packed A-fragment layout (per-row K-chunk swizzle)
__global__ void k_pack_a_bf16(const float* __restrict__ in,
                              unsigned short* __restrict__ out, long long nRows) {
    long long i = blockIdx.x * (long long)blockDim.x + threadIdx.x;
    if (i >= nRows * KDIM) return;
    long long row = i >> 8;
    int col = (int)(i & 255);
    int kc = col >> 5, r = col & 31;
    out[(row << 8) + (kc << 5) + apos(r)] = f32_to_bf16_rne(in[i]);
}

// f32 W[256,Ncols] -> bf16 packed B layout: [kchunk][n][32 K-halves contiguous]
__global__ void k_pack_w_bf16(const float* __restrict__ in,
                              unsigned short* __restrict__ out, int Ncols) {
    int i = blockIdx.x * blockDim.x + threadIdx.x;
    if (i >= KDIM * Ncols) return;
    int k = i / Ncols, n = i - k * Ncols;
    int kc = k >> 5, r = k & 31;
    out[((size_t)(kc * Ncols + n) << 5) + apos(r)] = f32_to_bf16_rne(in[i]);
}

// C[M,Ncols] = Ap[M,256](packed bf16) * Bp(packed bf16).
// One wave per 16x32 output slab: A fragment reused across 2 WMMA tiles,
// K=256 fully unrolled -> 16 v_wmma_f32_16x16x32_bf16 back-to-back.
__global__ void k_gemm_wmma_bf16(const unsigned short* __restrict__ Ap,
                                 const unsigned short* __restrict__ Bp,
                                 float* __restrict__ C,
                                 int Ncols, int nGroups) {
    int gwave = (int)((blockIdx.x * (long long)blockDim.x + threadIdx.x) >> 5);
    int lane  = threadIdx.x & 31;
    if (gwave >= nGroups) return;                 // wave-uniform
    int gPerRow = Ncols >> 5;
    int tm = gwave / gPerRow;
    int tg = gwave - tm * gPerRow;
    int half = lane >> 4;
    int l15  = lane & 15;
    const unsigned short* arow = Ap + ((size_t)(tm * 16 + l15)) * KDIM;
    int n0 = tg * 32 + l15;                       // col of tile 0 (tile 1 = +16)
    v8f acc0 = {}, acc1 = {};
#pragma unroll
    for (int kc = 0; kc < 8; ++kc) {
        v16bf af  = *reinterpret_cast<const v16bf*>(arow + (kc << 5) + half * 16);
        v16bf bf0 = *reinterpret_cast<const v16bf*>(
                        Bp + (((size_t)kc * Ncols + n0) << 5) + half * 16);
        v16bf bf1 = *reinterpret_cast<const v16bf*>(
                        Bp + (((size_t)kc * Ncols + n0 + 16) << 5) + half * 16);
        acc0 = __builtin_amdgcn_wmma_f32_16x16x32_bf16(
                   false, af, false, bf0, (short)0, acc0, false, false);
        acc1 = __builtin_amdgcn_wmma_f32_16x16x32_bf16(
                   false, af, false, bf1, (short)0, acc1, false, false);
    }
    int rbase = tm * 16 + half * 8;               // C/D: VGPR r -> M=r / M=r+8
    int c0 = tg * 32 + l15, c1 = c0 + 16;
#pragma unroll
    for (int r = 0; r < 8; ++r) {
        C[(size_t)(rbase + r) * Ncols + c0] = acc0[r];
        C[(size_t)(rbase + r) * Ncols + c1] = acc1[r];
    }
}

// per-(node,head) attention dots: a_s = <h[n,hd,:],att_src[hd]>, a_d likewise
__global__ void k_att_scores(const float* __restrict__ h,
                             const float* __restrict__ att_s,
                             const float* __restrict__ att_d,
                             float* __restrict__ as_, float* __restrict__ ad_,
                             int nH, int C) {
    int idx = blockIdx.x * blockDim.x + threadIdx.x;
    if (idx >= NNODES * nH) return;
    int node = idx / nH, hd = idx - node * nH;
    const float* hp = h + (size_t)node * nH * C + (size_t)hd * C;
    const float* sp = att_s + hd * C;
    const float* dp = att_d + hd * C;
    float s = 0.f, d = 0.f;
    for (int c = 0; c < C; ++c) {
        float v = hp[c];
        s = fmaf(v, sp[c], s);
        d = fmaf(v, dp[c], d);
    }
    as_[idx] = s; ad_[idx] = d;
}

// edge_index is int64 per the reference (jnp int64): read as long long
__global__ void k_edge_max(const long long* __restrict__ ei,
                           const float* __restrict__ as_,
                           const float* __restrict__ ad_,
                           unsigned int* __restrict__ mkey, int nH) {
    long long idx = blockIdx.x * (long long)blockDim.x + threadIdx.x;
    if (idx >= (long long)E2TOT * nH) return;
    long long e = idx / nH; int hd = (int)(idx - e * nH);
    long long s = (e < NEDGES) ? ei[e]          : (e - NEDGES);
    long long d = (e < NEDGES) ? ei[NEDGES + e] : (e - NEDGES);
    float v = as_[s * nH + hd] + ad_[d * nH + hd];
    v = (v < 0.f) ? 0.2f * v : v;                 // leaky_relu(0.2)
    atomicMax(&mkey[d * nH + hd], fkey(v));
}

__global__ void k_edge_sum(const long long* __restrict__ ei,
                           const float* __restrict__ as_,
                           const float* __restrict__ ad_,
                           const unsigned int* __restrict__ mkey,
                           float* __restrict__ ssum, int nH) {
    long long idx = blockIdx.x * (long long)blockDim.x + threadIdx.x;
    if (idx >= (long long)E2TOT * nH) return;
    long long e = idx / nH; int hd = (int)(idx - e * nH);
    long long s = (e < NEDGES) ? ei[e]          : (e - NEDGES);
    long long d = (e < NEDGES) ? ei[NEDGES + e] : (e - NEDGES);
    float v = as_[s * nH + hd] + ad_[d * nH + hd];
    v = (v < 0.f) ? 0.2f * v : v;
    float ex = expf(v - fkey_dec(mkey[d * nH + hd]));
    atomicAdd(&ssum[d * nH + hd], ex);
}

// one wave32 per edge: out[dst,f] += alpha(edge,head(f)) * h[src,f]
__global__ void k_edge_agg(const long long* __restrict__ ei,
                           const float* __restrict__ as_,
                           const float* __restrict__ ad_,
                           const unsigned int* __restrict__ mkey,
                           const float* __restrict__ ssum,
                           const float* __restrict__ h,
                           float* __restrict__ agg, int nH, int C) {
    long long wid = (blockIdx.x * (long long)blockDim.x + threadIdx.x) >> 5;
    int lane = threadIdx.x & 31;
    if (wid >= (long long)E2TOT) return;
    long long s = (wid < NEDGES) ? ei[wid]          : (wid - NEDGES);
    long long d = (wid < NEDGES) ? ei[NEDGES + wid] : (wid - NEDGES);
    int F = nH * C;
    for (int f = lane; f < F; f += 32) {
        int hd = f / C;
        float v = as_[s * nH + hd] + ad_[d * nH + hd];
        v = (v < 0.f) ? 0.2f * v : v;
        float alpha = expf(v - fkey_dec(mkey[d * nH + hd])) / ssum[d * nH + hd];
        atomicAdd(&agg[d * (size_t)F + f], alpha * h[s * (size_t)F + f]);
    }
}

// bias + ELU, re-emit bf16 in packed A-fragment layout (F == 256 here)
__global__ void k_bias_elu_pack(const float* __restrict__ in,
                                const float* __restrict__ bias,
                                unsigned short* __restrict__ out) {
    long long i = blockIdx.x * (long long)blockDim.x + threadIdx.x;
    if (i >= (long long)NNODES * KDIM) return;
    long long node = i >> 8;
    int f = (int)(i & 255);
    float v = in[i] + bias[f];
    v = (v > 0.f) ? v : (expf(v) - 1.f);          // ELU
    int kc = f >> 5, r = f & 31;
    out[(node << 8) + (kc << 5) + apos(r)] = f32_to_bf16_rne(v);
}

__global__ void k_bias_f32(const float* __restrict__ in,
                           const float* __restrict__ bias,
                           float* __restrict__ out, int F) {
    long long i = blockIdx.x * (long long)blockDim.x + threadIdx.x;
    if (i >= (long long)NNODES * F) return;
    out[i] = in[i] + bias[i % F];
}

__global__ void k_pool(const float* __restrict__ h3,
                       const long long* __restrict__ batch,
                       float* __restrict__ sums, float* __restrict__ cnts) {
    int idx = blockIdx.x * blockDim.x + threadIdx.x;    // NNODES*32
    if (idx >= NNODES * 32) return;
    int node = idx >> 5, c = idx & 31;
    long long g = batch[node];
    atomicAdd(&sums[g * 32 + c], h3[idx]);
    if (c == 0) atomicAdd(&cnts[g], 1.0f);
}

__global__ void k_final(const float* __restrict__ sums,
                        const float* __restrict__ cnts,
                        const float* __restrict__ linW,
                        const float* __restrict__ linb,
                        float* __restrict__ out) {
    int g = blockIdx.x * blockDim.x + threadIdx.x;
    if (g >= NGRAPH) return;
    float cnt = fmaxf(cnts[g], 1.0f);
    float acc = 0.f;
    for (int c = 0; c < 32; ++c)
        acc = fmaf(sums[g * 32 + c] / cnt, linW[c], acc);
    out[g] = acc + linb[0];
}

// ---------------- host side ----------------
static inline unsigned nb(long long n, int t) { return (unsigned)((n + t - 1) / t); }
static inline size_t al256(size_t x) { return (x + 255) & ~(size_t)255; }

static void run_gat_layer(const unsigned short* xb, const float* W,
                          const float* att_s, const float* att_d, const float* bias,
                          unsigned short* Wb, float* h, float* agg,
                          float* as_, float* ad_, unsigned int* mkey, float* ssum,
                          const long long* ei, int nH, int C,
                          unsigned short* out_bf, float* out_f32, hipStream_t st) {
    const int F = nH * C;
    // weight -> packed bf16 B layout
    k_pack_w_bf16<<<nb((long long)KDIM * F, 256), 256, 0, st>>>(W, Wb, F);
    // h = xb @ Wb  (WMMA bf16 -> f32), one wave per 16x32 slab
    int nGroups = (NNODES / 16) * (F / 32);           // 50000 % 16 == 0, F % 32 == 0
    k_gemm_wmma_bf16<<<nb((long long)nGroups * 32, 128), 128, 0, st>>>(
        xb, Wb, h, F, nGroups);
    // attention logits per (node, head)
    k_att_scores<<<nb((long long)NNODES * nH, 256), 256, 0, st>>>(
        h, att_s, att_d, as_, ad_, nH, C);
    // reset softmax stats + aggregation buffer
    k_zero<<<nb((long long)NNODES * nH, 256), 256, 0, st>>>(mkey, (long long)NNODES * nH);
    k_zero<<<nb((long long)NNODES * nH, 256), 256, 0, st>>>((unsigned int*)ssum,
                                                            (long long)NNODES * nH);
    k_zero<<<nb((long long)NNODES * F, 256), 256, 0, st>>>((unsigned int*)agg,
                                                           (long long)NNODES * F);
    // edge sweeps: segment max, segment sum, weighted scatter-add
    k_edge_max<<<nb((long long)E2TOT * nH, 256), 256, 0, st>>>(ei, as_, ad_, mkey, nH);
    k_edge_sum<<<nb((long long)E2TOT * nH, 256), 256, 0, st>>>(ei, as_, ad_, mkey, ssum, nH);
    k_edge_agg<<<nb((long long)E2TOT * 32, 256), 256, 0, st>>>(
        ei, as_, ad_, mkey, ssum, h, agg, nH, C);
    // epilogue
    if (out_bf)
        k_bias_elu_pack<<<nb((long long)NNODES * KDIM, 256), 256, 0, st>>>(agg, bias, out_bf);
    else
        k_bias_f32<<<nb((long long)NNODES * F, 256), 256, 0, st>>>(agg, bias, out_f32, F);
}

extern "C" void kernel_launch(void* const* d_in, const int* in_sizes, int n_in,
                              void* d_out, int out_size, void* d_ws, size_t ws_size,
                              hipStream_t stream) {
    (void)in_sizes; (void)n_in; (void)out_size; (void)ws_size;
    const float*     x     = (const float*)d_in[0];
    const long long* ei    = (const long long*)d_in[1];   // int64 in reference
    const long long* batch = (const long long*)d_in[2];
    const float* W1 = (const float*)d_in[3];
    const float* as1 = (const float*)d_in[4];
    const float* ad1 = (const float*)d_in[5];
    const float* b1 = (const float*)d_in[6];
    const float* W2 = (const float*)d_in[7];
    const float* as2 = (const float*)d_in[8];
    const float* ad2 = (const float*)d_in[9];
    const float* b2 = (const float*)d_in[10];
    const float* W3 = (const float*)d_in[11];
    const float* as3 = (const float*)d_in[12];
    const float* ad3 = (const float*)d_in[13];
    const float* b3 = (const float*)d_in[14];
    const float* linW = (const float*)d_in[15];
    const float* linb = (const float*)d_in[16];
    float* out = (float*)d_out;

    // workspace carve-up (~135 MB)
    char* w = (char*)d_ws;
    unsigned short* xb  = (unsigned short*)w; w += al256((size_t)NNODES * KDIM * 2);
    unsigned short* Wb  = (unsigned short*)w; w += al256((size_t)KDIM * 256 * 2);
    float* h    = (float*)w;        w += al256((size_t)NNODES * 256 * 4);
    float* agg  = (float*)w;        w += al256((size_t)NNODES * 256 * 4);
    float* asx  = (float*)w;        w += al256((size_t)NNODES * 8 * 4);
    float* adx  = (float*)w;        w += al256((size_t)NNODES * 8 * 4);
    unsigned int* mkey = (unsigned int*)w; w += al256((size_t)NNODES * 8 * 4);
    float* ssum = (float*)w;        w += al256((size_t)NNODES * 8 * 4);
    float* sums = (float*)w;        w += al256((size_t)NGRAPH * 32 * 4);
    float* cnts = (float*)w;        w += al256((size_t)NGRAPH * 4);

    // input features -> packed bf16 A layout
    k_pack_a_bf16<<<nb((long long)NNODES * KDIM, 256), 256, 0, stream>>>(
        x, xb, NNODES);

    // layer 1: heads=8, C=32, concat, ELU  -> xb (packed bf16)
    run_gat_layer(xb, W1, as1, ad1, b1, Wb, h, agg, asx, adx, mkey, ssum,
                  ei, 8, 32, xb, nullptr, stream);
    // layer 2: same -> xb (packed bf16)
    run_gat_layer(xb, W2, as2, ad2, b2, Wb, h, agg, asx, adx, mkey, ssum,
                  ei, 8, 32, xb, nullptr, stream);
    // layer 3: heads=1, C=32, mean-over-1-head == identity, bias only -> h (f32)
    run_gat_layer(xb, W3, as3, ad3, b3, Wb, h, agg, asx, adx, mkey, ssum,
                  ei, 1, 32, nullptr, h, stream);

    // global mean pool per graph + linear head
    k_zero<<<nb(NGRAPH * 32, 256), 256, 0, stream>>>((unsigned int*)sums, NGRAPH * 32);
    k_zero<<<nb(NGRAPH, 256), 256, 0, stream>>>((unsigned int*)cnts, NGRAPH);
    k_pool<<<nb((long long)NNODES * 32, 256), 256, 0, stream>>>(h, batch, sums, cnts);
    k_final<<<nb(NGRAPH, 64), 64, 0, stream>>>(sums, cnts, linW, linb, out);
}
